// LoRA_MOE_FFN_28381143892015
// MI455X (gfx1250) — matrix operand order repaired and checked by reading, not compiled, not run on hardware
//
#include <hip/hip_runtime.h>

#define TOK 4096
#define DM  2048
#define DH  8192
#define NE  8
#define NR  16
#define SCAL 2.0f
#define BM 128
#define BN 128
#define KB 32
#define LSTR 40   // LDS row stride in halves: 32 data + 8 pad (80B rows, 16B-aligned chunks)

typedef __attribute__((ext_vector_type(16))) __bf16 v16bf;
typedef __attribute__((ext_vector_type(8)))  float v8f;
typedef __attribute__((ext_vector_type(2)))  float v2f;
typedef __attribute__((ext_vector_type(2)))  __bf16 v2bf;

union FragBF { v16bf v; uint4 q[2]; };

// Hardware bf16 conversion (RNE): lowers to v_cvt_pk_bf16_f32-class ops.
__device__ __forceinline__ unsigned pack2(float a, float b) {
  v2f f = {a, b};
  v2bf r = __builtin_convertvector(f, v2bf);
  return __builtin_bit_cast(unsigned, r);
}
__device__ __forceinline__ unsigned short f2bf(float f) {
  __bf16 r = (__bf16)f;
  return __builtin_bit_cast(unsigned short, r);
}
__device__ __forceinline__ float bf2f(unsigned short h) {
  return __uint_as_float(((unsigned)h) << 16);
}
__device__ __forceinline__ uint4 pack8(float4 a, float4 b) {
  uint4 t;
  t.x = pack2(a.x, a.y); t.y = pack2(a.z, a.w);
  t.z = pack2(b.x, b.y); t.w = pack2(b.z, b.w);
  return t;
}
__device__ __forceinline__ float wave_sum(float v) {
#pragma unroll
  for (int off = 16; off > 0; off >>= 1) v += __shfl_xor(v, off, 32);
  return v;
}

// ---------------------------------------------------------------------------
// Kernel 1: router softmax/argmax + t_gate/t_up for selected expert.
// One wave (32 lanes) per token; 8 waves per 256-thread block.
// ---------------------------------------------------------------------------
__global__ __launch_bounds__(256) void k_router(
    const float* __restrict__ x, const float* __restrict__ rw,
    const float* __restrict__ rb,
    const float* __restrict__ Ag, const float* __restrict__ Au,
    float* __restrict__ tg, float* __restrict__ tu,
    int* __restrict__ idx, float* __restrict__ routing, float* __restrict__ ec)
{
  int lane  = threadIdx.x & 31;
  int token = blockIdx.x * 8 + (threadIdx.x >> 5);
  const float* xp = x + (size_t)token * DM;

  float xr[64];
#pragma unroll
  for (int j = 0; j < 16; ++j) {
    float4 v = *(const float4*)(xp + j * 128 + lane * 4);
    xr[j*4+0] = v.x; xr[j*4+1] = v.y; xr[j*4+2] = v.z; xr[j*4+3] = v.w;
  }

  float logit[NE];
#pragma unroll
  for (int e = 0; e < NE; ++e) {
    const float* rp = rw + e * DM;
    float s = 0.f;
#pragma unroll
    for (int j = 0; j < 16; ++j) {
      float4 v = *(const float4*)(rp + j * 128 + lane * 4);
      s += xr[j*4+0]*v.x + xr[j*4+1]*v.y + xr[j*4+2]*v.z + xr[j*4+3]*v.w;
    }
    logit[e] = wave_sum(s) + rb[e];
  }

  float mx = logit[0]; int am = 0;
#pragma unroll
  for (int e = 1; e < NE; ++e) if (logit[e] > mx) { mx = logit[e]; am = e; }
  float p[NE]; float den = 0.f;
#pragma unroll
  for (int e = 0; e < NE; ++e) { p[e] = __expf(logit[e] - mx); den += p[e]; }
  float inv = 1.f / den;
  if (lane == 0) {
#pragma unroll
    for (int e = 0; e < NE; ++e) {
      routing[token*NE + e] = p[e] * inv;
      ec[token*NE + e]      = (e == am) ? 1.f : 0.f;
    }
    idx[token] = am;
  }

  for (int pr = 0; pr < 2; ++pr) {
    const float* A   = (pr == 0 ? Ag : Au) + (size_t)am * NR * DM;
    float*       dst = (pr == 0 ? tg : tu) + token * NR;
#pragma unroll
    for (int r = 0; r < NR; ++r) {
      const float* ap = A + r * DM;
      float s = 0.f;
#pragma unroll
      for (int j = 0; j < 16; ++j) {
        float4 v = *(const float4*)(ap + j * 128 + lane * 4);
        s += xr[j*4+0]*v.x + xr[j*4+1]*v.y + xr[j*4+2]*v.z + xr[j*4+3]*v.w;
      }
      s = wave_sum(s);
      if (lane == 0) dst[r] = s;
    }
  }
}

// ---------------------------------------------------------------------------
// Kernel 2: fused gate/up GEMM (bf16 WMMA, f32 accum) + LoRA epilogue + SiLU.
// 512 threads = 16 waves in 4x4 grid; per-wave tile 32x32; double-buffered LDS.
// ---------------------------------------------------------------------------
__global__ __launch_bounds__(512) void k_gateup(
    const float* __restrict__ x,  const float* __restrict__ Wg,
    const float* __restrict__ Wu, const float* __restrict__ Bg,
    const float* __restrict__ Bu, const float* __restrict__ tg,
    const float* __restrict__ tu, const int* __restrict__ idx,
    unsigned short* __restrict__ h)
{
  __shared__ unsigned short Xs [2][BM * LSTR];
  __shared__ unsigned short Wgs[2][BN * LSTR];
  __shared__ unsigned short Wus[2][BN * LSTR];
  __shared__ float tgs[BM * NR];
  __shared__ float tus[BM * NR];
  __shared__ int   es [BM];

  const int tid  = threadIdx.x;
  const int lane = tid & 31, wave = tid >> 5;
  const int wm = wave & 3, wn = wave >> 2;           // 4x4 wave grid, 32x32 tiles
  const int m0 = blockIdx.y * BM, n0 = blockIdx.x * BN;

  if (tid < BM) es[tid] = idx[m0 + tid];
  for (int i = tid; i < BM * NR; i += 512) {
    tgs[i] = tg[(size_t)m0 * NR + i];
    tus[i] = tu[(size_t)m0 * NR + i];
  }

  v8f zero8 = {0.f,0.f,0.f,0.f,0.f,0.f,0.f,0.f};
  v8f accg[2][2], accu[2][2];
#pragma unroll
  for (int mi = 0; mi < 2; ++mi)
#pragma unroll
    for (int ni = 0; ni < 2; ++ni) { accg[mi][ni] = zero8; accu[mi][ni] = zero8; }

  // Loader: 512 threads, each does 8 f32 (2 float4) per matrix per K-step.
  const int lrow = tid >> 2, lq = tid & 3;           // row 0..127, 8-wide k octant
  const float* xrow = x  + (size_t)(m0 + lrow) * DM + lq * 8;
  const float* grow = Wg + (size_t)(n0 + lrow) * DM + lq * 8;
  const float* urow = Wu + (size_t)(n0 + lrow) * DM + lq * 8;
  const int dst = lrow * LSTR + lq * 8;              // halves offset in LDS row

  float4 rx0, rx1, rg0, rg1, ru0, ru1;
  rx0 = *(const float4*)(xrow);     rx1 = *(const float4*)(xrow + 4);
  rg0 = *(const float4*)(grow);     rg1 = *(const float4*)(grow + 4);
  ru0 = *(const float4*)(urow);     ru1 = *(const float4*)(urow + 4);
  *(uint4*)(&Xs [0][dst]) = pack8(rx0, rx1);
  *(uint4*)(&Wgs[0][dst]) = pack8(rg0, rg1);
  *(uint4*)(&Wus[0][dst]) = pack8(ru0, ru1);
  __syncthreads();

  const int NKB = DM / KB;                           // 64 K-steps
  const int ac0 = (lane >> 4) * 16;                  // A frag chunk base (halves)
  const int bc0 = (lane >> 4) * 16;                  // B frag chunk base (halves)
  const int fr  = lane & 15;

  for (int it = 0; it < NKB; ++it) {
    const int cur = it & 1;
    const bool more = (it + 1) < NKB;
    if (more) {
      const int kb = (it + 1) * KB;
      rx0 = *(const float4*)(xrow + kb); rx1 = *(const float4*)(xrow + kb + 4);
      rg0 = *(const float4*)(grow + kb); rg1 = *(const float4*)(grow + kb + 4);
      ru0 = *(const float4*)(urow + kb); ru1 = *(const float4*)(urow + kb + 4);
    }

    FragBF a[2], bg[2], bu[2];
#pragma unroll
    for (int mi = 0; mi < 2; ++mi) {
      const unsigned short* p = &Xs[cur][(wm*32 + mi*16 + fr) * LSTR];
      a[mi].q[0] = *(const uint4*)(p + ac0);
      a[mi].q[1] = *(const uint4*)(p + ac0 + 8);
    }
#pragma unroll
    for (int ni = 0; ni < 2; ++ni) {
      const unsigned short* pg = &Wgs[cur][(wn*32 + ni*16 + fr) * LSTR];
      bg[ni].q[0] = *(const uint4*)(pg + bc0);
      bg[ni].q[1] = *(const uint4*)(pg + bc0 + 8);
      const unsigned short* pu = &Wus[cur][(wn*32 + ni*16 + fr) * LSTR];
      bu[ni].q[0] = *(const uint4*)(pu + bc0);
      bu[ni].q[1] = *(const uint4*)(pu + bc0 + 8);
    }
#pragma unroll
    for (int mi = 0; mi < 2; ++mi)
#pragma unroll
      for (int ni = 0; ni < 2; ++ni) {
        accg[mi][ni] = __builtin_amdgcn_wmma_f32_16x16x32_bf16(
            false, a[mi].v, false, bg[ni].v, (short)0, accg[mi][ni], false, false);
        accu[mi][ni] = __builtin_amdgcn_wmma_f32_16x16x32_bf16(
            false, a[mi].v, false, bu[ni].v, (short)0, accu[mi][ni], false, false);
      }

    if (more) {
      const int nxt = cur ^ 1;
      *(uint4*)(&Xs [nxt][dst]) = pack8(rx0, rx1);
      *(uint4*)(&Wgs[nxt][dst]) = pack8(rg0, rg1);
      *(uint4*)(&Wus[nxt][dst]) = pack8(ru0, ru1);
    }
    __syncthreads();
  }

  // Epilogue: + scaling * (t . B[e,n,:]) ; h = silu(g) * u -> bf16
  const int nl = lane & 15, half = lane >> 4;
#pragma unroll
  for (int mi = 0; mi < 2; ++mi)
#pragma unroll
    for (int ni = 0; ni < 2; ++ni) {
      const int ngl = n0 + wn*32 + ni*16 + nl;
#pragma unroll
      for (int r8 = 0; r8 < 8; ++r8) {
        const int ml = wm*32 + mi*16 + half*8 + r8;
        const int e  = es[ml];
        const float* bgr = Bg + ((size_t)e * DH + ngl) * NR;
        const float* bur = Bu + ((size_t)e * DH + ngl) * NR;
        const float* tgr = &tgs[ml * NR];
        const float* tur = &tus[ml * NR];
        float lg = 0.f, lu = 0.f;
#pragma unroll
        for (int r = 0; r < NR; ++r) { lg += tgr[r] * bgr[r]; lu += tur[r] * bur[r]; }
        float g = accg[mi][ni][r8] + SCAL * lg;
        float u = accu[mi][ni][r8] + SCAL * lu;
        float s = g / (1.f + __expf(-g));
        h[(size_t)(m0 + ml) * DH + ngl] = f2bf(s * u);
      }
    }
}

// ---------------------------------------------------------------------------
// Kernel 3: t_down = h . A_down[e]^T   (one wave per token)
// ---------------------------------------------------------------------------
__global__ __launch_bounds__(256) void k_tdown(
    const unsigned short* __restrict__ h, const float* __restrict__ Ad,
    const int* __restrict__ idx, float* __restrict__ td)
{
  int lane  = threadIdx.x & 31;
  int token = blockIdx.x * 8 + (threadIdx.x >> 5);
  int e = idx[token];
  const unsigned short* hp = h + (size_t)token * DH;
  const float* A = Ad + (size_t)e * NR * DH;

  float acc[NR];
#pragma unroll
  for (int r = 0; r < NR; ++r) acc[r] = 0.f;

  for (int j = 0; j < DH / 32; ++j) {
    int k = j * 32 + lane;
    float hv = bf2f(hp[k]);
#pragma unroll
    for (int r = 0; r < NR; ++r) acc[r] += hv * A[r * DH + k];
  }
#pragma unroll
  for (int r = 0; r < NR; ++r) {
    float s = wave_sum(acc[r]);
    if (lane == 0) td[token * NR + r] = s;
  }
}

// ---------------------------------------------------------------------------
// Kernel 4: down GEMM out = h @ Wd^T + LoRA-down (f32 out)
// 512 threads, 4x4 wave grid, 32x32 per-wave tile, double-buffered LDS.
// ---------------------------------------------------------------------------
__global__ __launch_bounds__(512) void k_down(
    const unsigned short* __restrict__ h, const float* __restrict__ Wd,
    const float* __restrict__ Bd, const float* __restrict__ td,
    const int* __restrict__ idx, float* __restrict__ out)
{
  __shared__ unsigned short Hs [2][BM * LSTR];
  __shared__ unsigned short Wds[2][BN * LSTR];
  __shared__ float tds[BM * NR];
  __shared__ int   es [BM];

  const int tid  = threadIdx.x;
  const int lane = tid & 31, wave = tid >> 5;
  const int wm = wave & 3, wn = wave >> 2;
  const int m0 = blockIdx.y * BM, n0 = blockIdx.x * BN;

  if (tid < BM) es[tid] = idx[m0 + tid];
  for (int i = tid; i < BM * NR; i += 512) tds[i] = td[(size_t)m0 * NR + i];

  v8f zero8 = {0.f,0.f,0.f,0.f,0.f,0.f,0.f,0.f};
  v8f acc[2][2];
#pragma unroll
  for (int mi = 0; mi < 2; ++mi)
#pragma unroll
    for (int ni = 0; ni < 2; ++ni) acc[mi][ni] = zero8;

  const int lrow = tid >> 2, lq = tid & 3;
  const unsigned short* hrow = h + (size_t)(m0 + lrow) * DH + lq * 8;   // bf16 direct
  const float* drow = Wd + (size_t)(n0 + lrow) * DH + lq * 8;
  const int dst = lrow * LSTR + lq * 8;

  uint4 rh; float4 rd0, rd1;
  rh  = *(const uint4*)(hrow);
  rd0 = *(const float4*)(drow); rd1 = *(const float4*)(drow + 4);
  *(uint4*)(&Hs [0][dst]) = rh;
  *(uint4*)(&Wds[0][dst]) = pack8(rd0, rd1);
  __syncthreads();

  const int NKB = DH / KB;                           // 256 K-steps
  const int ac0 = (lane >> 4) * 16;
  const int bc0 = (lane >> 4) * 16;
  const int fr  = lane & 15;

  for (int it = 0; it < NKB; ++it) {
    const int cur = it & 1;
    const bool more = (it + 1) < NKB;
    if (more) {
      const int kb = (it + 1) * KB;
      rh  = *(const uint4*)(hrow + kb);
      rd0 = *(const float4*)(drow + kb); rd1 = *(const float4*)(drow + kb + 4);
    }

    FragBF a[2], b[2];
#pragma unroll
    for (int mi = 0; mi < 2; ++mi) {
      const unsigned short* p = &Hs[cur][(wm*32 + mi*16 + fr) * LSTR];
      a[mi].q[0] = *(const uint4*)(p + ac0);
      a[mi].q[1] = *(const uint4*)(p + ac0 + 8);
    }
#pragma unroll
    for (int ni = 0; ni < 2; ++ni) {
      const unsigned short* p = &Wds[cur][(wn*32 + ni*16 + fr) * LSTR];
      b[ni].q[0] = *(const uint4*)(p + bc0);
      b[ni].q[1] = *(const uint4*)(p + bc0 + 8);
    }
#pragma unroll
    for (int mi = 0; mi < 2; ++mi)
#pragma unroll
      for (int ni = 0; ni < 2; ++ni)
        acc[mi][ni] = __builtin_amdgcn_wmma_f32_16x16x32_bf16(
            false, a[mi].v, false, b[ni].v, (short)0, acc[mi][ni], false, false);

    if (more) {
      const int nxt = cur ^ 1;
      *(uint4*)(&Hs [nxt][dst]) = rh;
      *(uint4*)(&Wds[nxt][dst]) = pack8(rd0, rd1);
    }
    __syncthreads();
  }

  const int nl = lane & 15, half = lane >> 4;
#pragma unroll
  for (int mi = 0; mi < 2; ++mi)
#pragma unroll
    for (int ni = 0; ni < 2; ++ni) {
      const int ngl = n0 + wn*32 + ni*16 + nl;
#pragma unroll
      for (int r8 = 0; r8 < 8; ++r8) {
        const int ml = wm*32 + mi*16 + half*8 + r8;
        const int e  = es[ml];
        const float* br = Bd + ((size_t)e * DM + ngl) * NR;
        const float* tr = &tds[ml * NR];
        float lo = 0.f;
#pragma unroll
        for (int r = 0; r < NR; ++r) lo += tr[r] * br[r];
        out[(size_t)(m0 + ml) * DM + ngl] = acc[mi][ni][r8] + SCAL * lo;
      }
    }
}

// ---------------------------------------------------------------------------
extern "C" void kernel_launch(void* const* d_in, const int* in_sizes, int n_in,
                              void* d_out, int out_size, void* d_ws, size_t ws_size,
                              hipStream_t stream) {
  (void)in_sizes; (void)n_in; (void)out_size; (void)ws_size;
  const float* x  = (const float*)d_in[0];
  const float* wg = (const float*)d_in[1];
  const float* wu = (const float*)d_in[2];
  const float* wd = (const float*)d_in[3];
  const float* rw = (const float*)d_in[4];
  const float* rb = (const float*)d_in[5];
  const float* Ag = (const float*)d_in[6];
  const float* Bg = (const float*)d_in[7];
  const float* Au = (const float*)d_in[8];
  const float* Bu = (const float*)d_in[9];
  const float* Ad = (const float*)d_in[10];
  const float* Bd = (const float*)d_in[11];

  float* out     = (float*)d_out;
  float* routing = out + (size_t)TOK * DM;       // [4096,8]
  float* ec      = routing + (size_t)TOK * NE;   // [4096,8]

  char* ws = (char*)d_ws;
  unsigned short* h  = (unsigned short*)ws;      // 64 MB bf16
  float* tg  = (float*)(ws + (size_t)TOK * DH * 2);
  float* tu  = tg + (size_t)TOK * NR;
  float* td  = tu + (size_t)TOK * NR;
  int*   idx = (int*)(td + (size_t)TOK * NR);

  k_router<<<TOK / 8, 256, 0, stream>>>(x, rw, rb, Ag, Au, tg, tu, idx, routing, ec);
  k_gateup<<<dim3(DH / BN, TOK / BM), 512, 0, stream>>>(x, wg, wu, Bg, Bu, tg, tu, idx, h);
  k_tdown<<<TOK / 8, 256, 0, stream>>>(h, Ad, idx, td);
  k_down<<<dim3(DM / BN, TOK / BM), 512, 0, stream>>>(h, wd, Bd, td, idx, out);
}
